// TrustGNN_87625922773223
// MI455X (gfx1250) — compile-verified, matrix operand order checked by reading, not run on hardware
//
#include <hip/hip_runtime.h>
#include <math.h>

typedef __attribute__((ext_vector_type(16))) _Float16 v16h;
typedef __attribute__((ext_vector_type(8)))  _Float16 v8h;
typedef __attribute__((ext_vector_type(8)))  float    v8f;

#define GN 50000
#define GE 500000
#define GK 385
#define WTK 416              // K padded to 13*32 for the packed f16 weights

// ---------------------------------------------------------------------------
// float atomic max via int/uint ordering trick (init with -inf)
__device__ inline void atomicMaxF(float* addr, float v) {
    if (v >= 0.0f) atomicMax((int*)addr, __float_as_int(v));
    else           atomicMin((unsigned int*)addr, __float_as_uint(v));
}

// ---------------------------------------------------------------------------
__global__ void fill_f32(float* __restrict__ p, float v, int n) {
    int i = blockIdx.x * blockDim.x + threadIdx.x;
    if (i < n) p[i] = v;
}

// ---------------------------------------------------------------------------
// Pack We[K,OCsrc] (f32, row-major) into Wt[col][k] f16, zero-padded to WTK.
__global__ void pack_wt(const float* __restrict__ We, int OCsrc,
                        _Float16* __restrict__ Wt, int colOff) {
    int i = blockIdx.x * blockDim.x + threadIdx.x;   // i = c*WTK + k
    if (i >= OCsrc * WTK) return;
    int c = i / WTK, k = i - c * WTK;
    float v = (k < GK) ? We[(long long)k * OCsrc + c] : 0.0f;
    Wt[(long long)(colOff + c) * WTK + k] = (_Float16)v;
}

// ---------------------------------------------------------------------------
// xl = x @ Wl + bl ; xr = x @ Wr + br    (tiny GEMMs, VALU)
__global__ void node_linear(const float* __restrict__ x, int in_dim,
                            const float* __restrict__ Wl, const float* __restrict__ bl,
                            const float* __restrict__ Wr, const float* __restrict__ br,
                            float* __restrict__ xl, float* __restrict__ xr,
                            int n, int OC) {
    int idx = blockIdx.x * blockDim.x + threadIdx.x;
    if (idx >= n * OC) return;
    int nid = idx / OC, oc = idx - nid * OC;
    const float* xp = x + (long long)nid * in_dim;
    float sl = bl[oc], sr = br[oc];
    for (int i = 0; i < in_dim; ++i) {
        float xv = xp[i];
        sl += xv * Wl[i * OC + oc];
        sr += xv * Wr[i * OC + oc];
    }
    xl[idx] = sl;
    xr[idx] = sr;
}

// ---------------------------------------------------------------------------
// C[M, ldC] (cols [0,NT*16)) = A[M,385] @ Wt^T  -- WMMA f16->f32.
// A staged in LDS (f32->f16); B fragments read directly from packed f16 Wt
// (93KB max, cache resident). 8 waves x 16 rows = 128 rows / block.
template <int NT>
__global__ __launch_bounds__(256)
void ea_gemm(const float* __restrict__ A, const _Float16* __restrict__ Wt,
             float* __restrict__ C, int M, int ldC) {
    __shared__ _Float16 lA[128][40];            // 128 rows x 32 K (padded)

    const int tid  = threadIdx.x;
    const int lane = tid & 31;
    const int wave = tid >> 5;                  // 8 waves -> 128 rows / block
    const int rowBase = blockIdx.x * 128;
    const bool full = (rowBase + 128 <= M);

    v8f zero = {0.f, 0.f, 0.f, 0.f, 0.f, 0.f, 0.f, 0.f};
    v8f acc[NT];
#pragma unroll
    for (int t = 0; t < NT; ++t) acc[t] = zero;

    const int kb = (lane & 16) >> 1;            // 0 for lanes 0-15, 8 for 16-31
    const int rl = wave * 16 + (lane & 15);     // A-fragment row in LDS

    for (int k0 = 0; k0 < WTK; k0 += 32) {
        // ---- stage A tile --------------------------------------------------
        if (full && (k0 + 32 <= GK)) {
            // unguarded: 4 chunks of 4 contiguous K per thread, NT loads
#pragma unroll
            for (int i = tid * 4; i < 128 * 32; i += 1024) {
                int r = i >> 5, kk = i & 31;
                const float* p = A + (long long)(rowBase + r) * GK + k0 + kk;
                float x0 = __builtin_nontemporal_load(p + 0);
                float x1 = __builtin_nontemporal_load(p + 1);
                float x2 = __builtin_nontemporal_load(p + 2);
                float x3 = __builtin_nontemporal_load(p + 3);
                lA[r][kk + 0] = (_Float16)x0;
                lA[r][kk + 1] = (_Float16)x1;
                lA[r][kk + 2] = (_Float16)x2;
                lA[r][kk + 3] = (_Float16)x3;
            }
        } else {
            for (int i = tid; i < 128 * 32; i += 256) {
                int r = i >> 5, kk = i & 31;
                int gr = rowBase + r, gk = k0 + kk;
                float v = (gr < M && gk < GK) ? A[(long long)gr * GK + gk] : 0.0f;
                lA[r][kk] = (_Float16)v;
            }
        }
        __syncthreads();

        // ---- fragments + WMMA ---------------------------------------------
        // A fragment: element e -> K = kb + (e&7) + ((e>>3)<<4)
        v16h a;
#pragma unroll
        for (int e = 0; e < 16; ++e) a[e] = lA[rl][kb + (e & 7) + ((e >> 3) << 4)];

#pragma unroll
        for (int t = 0; t < NT; ++t) {
            int cl = t * 16 + (lane & 15);
            const _Float16* wp = Wt + (long long)cl * WTK + k0;
            v8h b0 = *(const v8h*)(wp + kb);        // K = kb .. kb+7   (16B aligned)
            v8h b1 = *(const v8h*)(wp + kb + 16);   // K = kb+16..kb+23 (16B aligned)
            v16h b;
#pragma unroll
            for (int e = 0; e < 8; ++e) { b[e] = b0[e]; b[e + 8] = b1[e]; }
            acc[t] = __builtin_amdgcn_wmma_f32_16x16x32_f16(
                false, a, false, b, (short)0, acc[t], false, false);
        }
        __syncthreads();
    }

    // ---- store: D vgpr i -> row = i + 8*(lane>=16), col = lane&15 ----------
    const int mhalf = (lane >> 4) << 3;
#pragma unroll
    for (int t = 0; t < NT; ++t) {
        int col = t * 16 + (lane & 15);
#pragma unroll
        for (int i = 0; i < 8; ++i) {
            int gr = rowBase + wave * 16 + i + mhalf;
            if (gr < M) C[(long long)gr * ldC + col] = acc[t][i];
        }
    }
}

// ---------------------------------------------------------------------------
// per-edge attention logits + segment max into mb[dst]
__global__ void edge_logits(const int* __restrict__ src, const int* __restrict__ dst,
                            const float* __restrict__ ea, int eaStride,
                            const float* __restrict__ xl, const float* __restrict__ xr,
                            const float* __restrict__ att,
                            float* __restrict__ lg, float* __restrict__ mb,
                            int ne, int H, int Cc) {
    int e = blockIdx.x * blockDim.x + threadIdx.x;
    if (e >= ne) return;
    int s = src[e], d = dst[e];
    int OC = H * Cc;
    const float* eap = ea + (long long)e * eaStride;
    const float* xjp = xl + (long long)s * OC;
    const float* xip = xr + (long long)d * OC;
    for (int h = 0; h < H; ++h) {
        float acc = 0.0f;
        for (int c = 0; c < Cc; ++c) {
            float v = xjp[h * Cc + c] + xip[h * Cc + c] + eap[h * Cc + c];
            v = v > 0.0f ? v : 0.2f * v;            // leaky_relu(0.2)
            acc += v * att[h * Cc + c];
        }
        lg[(long long)e * H + h] = acc;
        atomicMaxF(&mb[(long long)d * H + h], acc);
    }
}

// ---------------------------------------------------------------------------
// ex = exp(logit - m[dst]); segment sum into sb[dst]; ex overwrites lg
__global__ void edge_expsum(const int* __restrict__ dst,
                            float* __restrict__ lg, const float* __restrict__ mb,
                            float* __restrict__ sb, int ne, int H) {
    int i = blockIdx.x * blockDim.x + threadIdx.x;
    if (i >= ne * H) return;
    int e = i / H, h = i - e * H;
    int d = dst[e];
    float m = mb[(long long)d * H + h];
    if (!(m >= -3.0e38f && m <= 3.0e38f)) m = 0.0f;   // isfinite -> 0
    float ex = __expf(lg[i] - m);
    lg[i] = ex;
    atomicAdd(&sb[(long long)d * H + h], ex);
}

// ---------------------------------------------------------------------------
// out[dst] += xl[src] * alpha   (alpha = ex / (s[dst]+eps))
__global__ void edge_aggregate(const int* __restrict__ src, const int* __restrict__ dst,
                               const float* __restrict__ ex, const float* __restrict__ sb,
                               const float* __restrict__ xl, float* __restrict__ xa,
                               int ne, int H, int Cc) {
    int e = blockIdx.x * blockDim.x + threadIdx.x;
    if (e >= ne) return;
    int s = src[e], d = dst[e];
    int OC = H * Cc;
    for (int h = 0; h < H; ++h) {
        float alpha = ex[(long long)e * H + h] / (sb[(long long)d * H + h] + 1e-16f);
        for (int c = 0; c < Cc; ++c) {
            atomicAdd(&xa[(long long)d * OC + h * Cc + c],
                      xl[(long long)s * OC + h * Cc + c] * alpha);
        }
    }
}

// ---------------------------------------------------------------------------
__global__ void finalize_layer(const float* __restrict__ xa, const float* __restrict__ bias,
                               float* __restrict__ xo, int n, int OC, int do_elu) {
    int i = blockIdx.x * blockDim.x + threadIdx.x;
    if (i >= n * OC) return;
    int oc = i % OC;
    float v = xa[i] + bias[oc];
    if (do_elu) v = v > 0.0f ? v : (__expf(v) - 1.0f);
    xo[i] = v;
}

// ---------------------------------------------------------------------------
// edge MLP: concat(x3[src],x3[dst]) [32] -> 64 relu -> 32 relu -> 1
__global__ __launch_bounds__(256)
void edge_mlp(const int* __restrict__ src, const int* __restrict__ dst,
              const float* __restrict__ x3,
              const float* __restrict__ W1, const float* __restrict__ b1,
              const float* __restrict__ W2, const float* __restrict__ b2,
              const float* __restrict__ W3, const float* __restrict__ b3,
              float* __restrict__ out, int ne) {
    __shared__ float sW1[32 * 64];
    __shared__ float sW2[64 * 32];
    __shared__ float sW3[32];
    __shared__ float sb1[64];
    __shared__ float sb2[32];
    for (int i = threadIdx.x; i < 32 * 64; i += blockDim.x) sW1[i] = W1[i];
    for (int i = threadIdx.x; i < 64 * 32; i += blockDim.x) sW2[i] = W2[i];
    for (int i = threadIdx.x; i < 32;      i += blockDim.x) sW3[i] = W3[i];
    for (int i = threadIdx.x; i < 64;      i += blockDim.x) sb1[i] = b1[i];
    for (int i = threadIdx.x; i < 32;      i += blockDim.x) sb2[i] = b2[i];
    __syncthreads();

    int e = blockIdx.x * blockDim.x + threadIdx.x;
    if (e >= ne) return;
    int s = src[e], d = dst[e];
    float h0[32];
#pragma unroll
    for (int c = 0; c < 16; ++c) {
        h0[c]      = x3[(long long)s * 16 + c];
        h0[16 + c] = x3[(long long)d * 16 + c];
    }
    float h1[64];
    for (int o = 0; o < 64; ++o) {
        float a = sb1[o];
#pragma unroll
        for (int i = 0; i < 32; ++i) a += h0[i] * sW1[i * 64 + o];
        h1[o] = a > 0.0f ? a : 0.0f;
    }
    float ov = b3[0];
    for (int o = 0; o < 32; ++o) {
        float a = sb2[o];
#pragma unroll
        for (int i = 0; i < 64; ++i) a += h1[i] * sW2[i * 32 + o];
        a = a > 0.0f ? a : 0.0f;
        ov += a * sW3[o];
    }
    out[e] = ov;
}

// ---------------------------------------------------------------------------
extern "C" void kernel_launch(void* const* d_in, const int* in_sizes, int n_in,
                              void* d_out, int out_size, void* d_ws, size_t ws_size,
                              hipStream_t stream) {
    const int   N = GN, E = GE;
    const int*  eidx      = (const int*)d_in[1];
    const int*  src       = eidx;
    const int*  dst       = eidx + E;
    const float* edge_attr = (const float*)d_in[2];
    const float* node_emb  = (const float*)d_in[3];

    const float *l1_Wl=(const float*)d_in[4],  *l1_bl=(const float*)d_in[5];
    const float *l1_Wr=(const float*)d_in[6],  *l1_br=(const float*)d_in[7];
    const float *l1_We=(const float*)d_in[8],  *l1_att=(const float*)d_in[9],  *l1_bias=(const float*)d_in[10];
    const float *l2_Wl=(const float*)d_in[11], *l2_bl=(const float*)d_in[12];
    const float *l2_Wr=(const float*)d_in[13], *l2_br=(const float*)d_in[14];
    const float *l2_We=(const float*)d_in[15], *l2_att=(const float*)d_in[16], *l2_bias=(const float*)d_in[17];
    const float *l3_Wl=(const float*)d_in[18], *l3_bl=(const float*)d_in[19];
    const float *l3_Wr=(const float*)d_in[20], *l3_br=(const float*)d_in[21];
    const float *l3_We=(const float*)d_in[22], *l3_att=(const float*)d_in[23], *l3_bias=(const float*)d_in[24];
    const float *mW1=(const float*)d_in[25], *mb1=(const float*)d_in[26];
    const float *mW2=(const float*)d_in[27], *mb2=(const float*)d_in[28];
    const float *mW3=(const float*)d_in[29], *mb3=(const float*)d_in[30];

    // ---- workspace carve-up -------------------------------------------------
    // fused path: ea_all[E,112] + rest  (~285 MB); fallback: ea[E,64] + rest
    const long long restFloats = (long long)E * 4 + (long long)N * 4 * 2
                               + (long long)N * 64 * 4;
    const long long wtBytes    = (long long)112 * WTK * sizeof(_Float16);
    const long long needFused  = ((long long)E * 112 + restFloats) * 4 + wtBytes + 256;
    const bool fused = ((long long)ws_size >= needFused);
    const int  eaCols = fused ? 112 : 64;

    float* ws  = (float*)d_ws;
    float* ea  = ws;                              // E*eaCols
    float* lg  = ea + (long long)E * eaCols;      // E*4
    float* mb  = lg + (long long)E * 4;           // N*4
    float* sb  = mb + (long long)N * 4;           // N*4
    float* xl  = sb + (long long)N * 4;           // N*64
    float* xr  = xl + (long long)N * 64;          // N*64
    float* xa  = xr + (long long)N * 64;          // N*64 (layer accumulator)
    float* xc  = xa + (long long)N * 64;          // N*64 (layer input/output)
    _Float16* Wt = (_Float16*)(xc + (long long)N * 64);  // 112*WTK f16

    const int TB = 256;
    const int gblocks = (E + 127) / 128;
    const float NEG_INF = -__builtin_huge_valf();

    auto attention = [&](const float* xin, int in_dim, int H, int Cc,
                         const float* Wl, const float* bl, const float* Wr, const float* br,
                         const float* att, const float* bias,
                         const float* eaPtr, int eaStride,
                         float* xout, int do_elu) {
        int OC = H * Cc;
        node_linear<<<(N * OC + TB - 1) / TB, TB, 0, stream>>>(
            xin, in_dim, Wl, bl, Wr, br, xl, xr, N, OC);
        fill_f32<<<(N * H + TB - 1) / TB, TB, 0, stream>>>(mb, NEG_INF, N * H);
        fill_f32<<<(N * H + TB - 1) / TB, TB, 0, stream>>>(sb, 0.0f, N * H);
        fill_f32<<<(N * OC + TB - 1) / TB, TB, 0, stream>>>(xa, 0.0f, N * OC);
        edge_logits<<<(E + TB - 1) / TB, TB, 0, stream>>>(
            src, dst, eaPtr, eaStride, xl, xr, att, lg, mb, E, H, Cc);
        edge_expsum<<<(E * H + TB - 1) / TB, TB, 0, stream>>>(dst, lg, mb, sb, E, H);
        edge_aggregate<<<(E + TB - 1) / TB, TB, 0, stream>>>(
            src, dst, lg, sb, xl, xa, E, H, Cc);
        finalize_layer<<<(N * OC + TB - 1) / TB, TB, 0, stream>>>(
            xa, bias, xout, N, OC, do_elu);
    };

    if (fused) {
        // pack all three We into Wt[112][WTK] f16 and run ONE GEMM pass.
        pack_wt<<<(64 * WTK + TB - 1) / TB, TB, 0, stream>>>(l1_We, 64, Wt, 0);
        pack_wt<<<(32 * WTK + TB - 1) / TB, TB, 0, stream>>>(l2_We, 32, Wt, 64);
        pack_wt<<<(16 * WTK + TB - 1) / TB, TB, 0, stream>>>(l3_We, 16, Wt, 96);
        ea_gemm<7><<<gblocks, 256, 0, stream>>>(edge_attr, Wt, ea, E, 112);

        attention(node_emb, 32, 4, 16, l1_Wl, l1_bl, l1_Wr, l1_br, l1_att, l1_bias,
                  ea + 0, 112, xc, 1);
        attention(xc, 64, 2, 16, l2_Wl, l2_bl, l2_Wr, l2_br, l2_att, l2_bias,
                  ea + 64, 112, xc, 1);
        attention(xc, 32, 1, 16, l3_Wl, l3_bl, l3_Wr, l3_br, l3_att, l3_bias,
                  ea + 96, 112, xc, 0);
    } else {
        // per-layer GEMM into a shared ea[E,64] buffer
        pack_wt<<<(64 * WTK + TB - 1) / TB, TB, 0, stream>>>(l1_We, 64, Wt, 0);
        ea_gemm<4><<<gblocks, 256, 0, stream>>>(edge_attr, Wt, ea, E, 64);
        attention(node_emb, 32, 4, 16, l1_Wl, l1_bl, l1_Wr, l1_br, l1_att, l1_bias,
                  ea, 64, xc, 1);

        pack_wt<<<(32 * WTK + TB - 1) / TB, TB, 0, stream>>>(l2_We, 32, Wt, 0);
        ea_gemm<2><<<gblocks, 256, 0, stream>>>(edge_attr, Wt, ea, E, 32);
        attention(xc, 64, 2, 16, l2_Wl, l2_bl, l2_Wr, l2_br, l2_att, l2_bias,
                  ea, 32, xc, 1);

        pack_wt<<<(16 * WTK + TB - 1) / TB, TB, 0, stream>>>(l3_We, 16, Wt, 0);
        ea_gemm<1><<<gblocks, 256, 0, stream>>>(edge_attr, Wt, ea, E, 16);
        attention(xc, 32, 1, 16, l3_Wl, l3_bl, l3_Wr, l3_br, l3_att, l3_bias,
                  ea, 16, xc, 0);
    }

    // edge MLP -> d_out[E]
    edge_mlp<<<(E + TB - 1) / TB, TB, 0, stream>>>(
        src, dst, xc, mW1, mb1, mW2, mb2, mW3, mb3, (float*)d_out, E);
}
